// MoeLayer_76785425317967
// MI455X (gfx1250) — compile-verified, matrix operand order checked by reading, not run on hardware
//
#include <hip/hip_runtime.h>
#include <hip/hip_bf16.h>

typedef __bf16 bf16;
typedef __attribute__((ext_vector_type(16))) __bf16 v16bf;
typedef __attribute__((ext_vector_type(8)))  __bf16 bf16x8;
typedef __attribute__((ext_vector_type(4)))  __bf16 bf16x4;
typedef __attribute__((ext_vector_type(2)))  __bf16 bf16x2;
typedef __attribute__((ext_vector_type(8)))  float  v8f;
typedef __attribute__((ext_vector_type(2)))  float  f32x2;
typedef __attribute__((ext_vector_type(4)))  unsigned int u32x4;
typedef __attribute__((ext_vector_type(8)))  int    i32x8;
typedef __attribute__((ext_vector_type(4)))  int    i32x4;

#define D      1024
#define H      2730
#define HP     2752      // 43 * 64, zero-padded hidden
#define E      8
#define T      4096      // B*S tokens
#define CAP    4096      // max tokens per expert
#define ROWS   8192      // total assignments = T * TOP_K
#define MT     32        // M-tile rows per block

#if defined(__gfx1250__) && __has_builtin(__builtin_amdgcn_tensor_load_to_lds)
#define USE_TDM 1
#pragma message("CDNA5: TDM path ENABLED (tensor_load_to_lds builtin found)")
#else
#define USE_TDM 0
#if defined(__gfx1250__)
#pragma message("CDNA5: TDM path DISABLED (no tensor_load_to_lds builtin)")
#endif
#endif

// ---------------- routing ----------------
__global__ void moe_zero_cnt(int* cnt) {
    if (threadIdx.x < E) cnt[threadIdx.x] = 0;
}

__global__ __launch_bounds__(256)
void moe_gate(const float* __restrict__ x, const float* __restrict__ Wg,
              int* __restrict__ cnt, int* __restrict__ slot_list,
              float* __restrict__ gw) {
    const int t = blockIdx.x;
    const int tid = threadIdx.x;
    __shared__ float sm[E][256];
    float acc[E];
#pragma unroll
    for (int e = 0; e < E; ++e) acc[e] = 0.f;
    for (int d = tid; d < D; d += 256) {
        float xv = x[(size_t)t * D + d];
#pragma unroll
        for (int e = 0; e < E; ++e) acc[e] += xv * Wg[e * D + d];
    }
#pragma unroll
    for (int e = 0; e < E; ++e) sm[e][tid] = acc[e];
    __syncthreads();
    for (int s = 128; s > 0; s >>= 1) {
        if (tid < s)
#pragma unroll
            for (int e = 0; e < E; ++e) sm[e][tid] += sm[e][tid + s];
        __syncthreads();
    }
    if (tid == 0) {
        float lg[E];
#pragma unroll
        for (int e = 0; e < E; ++e) lg[e] = sm[e][0];
        int i0 = 0;
#pragma unroll
        for (int e = 1; e < E; ++e) if (lg[e] > lg[i0]) i0 = e;
        int i1 = (i0 == 0) ? 1 : 0;
#pragma unroll
        for (int e = 0; e < E; ++e) if (e != i0 && lg[e] > lg[i1]) i1 = e;
        float e1 = __expf(lg[i1] - lg[i0]);
        float inv = 1.f / (1.f + e1);
        int p0 = atomicAdd(&cnt[i0], 1);
        slot_list[i0 * CAP + p0] = t * 2 + 0;
        gw[t * 2 + 0] = inv;
        int p1 = atomicAdd(&cnt[i1], 1);
        slot_list[i1 * CAP + p1] = t * 2 + 1;
        gw[t * 2 + 1] = e1 * inv;
    }
}

__global__ void moe_scan(const int* __restrict__ cnt, int* __restrict__ off) {
    if (threadIdx.x == 0) {
        int s = 0;
        for (int e = 0; e < E; ++e) { off[e] = s; s += cnt[e]; }
    }
}

// ---------------- fragment helper ----------------
// 16-bit A/B fragment: element j of lane l -> K = (j/8)*16 + (l>=16?8:0) + j%8
__device__ inline v16bf load_frag(const bf16* __restrict__ row, int kbase) {
    v16bf r;
#pragma unroll
    for (int q = 0; q < 2; ++q)
#pragma unroll
        for (int j = 0; j < 8; ++j)
            r[q * 8 + j] = row[kbase + q * 16 + j];
    return r;
}

// pack float4 -> 4 bf16 (emits v_cvt_pk_bf16_f32 pairs)
__device__ inline bf16x4 pack4(float4 v) {
    bf16x4 p = {(bf16)v.x, (bf16)v.y, (bf16)v.z, (bf16)v.w};
    return p;
}

// ---------------- pass 1: hidden = silu(X W1^T) * (X W2^T) ----------------
// Block: 128 threads (4 waves). Tile: MT=32 rows x 64 hidden cols, K = 1024.
__global__ __launch_bounds__(128)
void moe_pass1(const float* __restrict__ x,
               const float* __restrict__ W1, const float* __restrict__ W2,
               const int* __restrict__ cnt, const int* __restrict__ off,
               const int* __restrict__ slot_list, bf16* __restrict__ hidden) {
    const int e  = blockIdx.z;
    const int mt = blockIdx.y;
    const int ht = blockIdx.x;          // 0..42 -> h0 = ht*64
    const int n  = cnt[e];
    if (mt * MT >= n) return;
    const int offE = off[e];

    __shared__ alignas(16) bf16 Xs[MT][64];         // 4 KB, per-K-chunk
    __shared__ alignas(16) bf16 Ws[2][4][16][64];   // 16 KB
    __shared__ int  toks[MT];

    const int tid  = threadIdx.x;
    const int lane = tid & 31;
    const int wave = tid >> 5;
    const int mrow = lane & 15;
    const int halfoff = (lane >> 4) << 3;   // 0 or 8

    if (tid < MT) {
        int pos = mt * MT + tid;
        toks[tid] = (pos < n) ? (slot_list[e * CAP + pos] >> 1) : 0;
    }
    __syncthreads();

    const int h0 = ht * 64 + wave * 16;
    const float* W1e = W1 + (size_t)e * H * D;
    const float* W2e = W2 + (size_t)e * H * D;

    v8f c1lo = {}, c1hi = {}, c2lo = {}, c2hi = {};
    const int wr = lane >> 1;
    const int wc = (lane & 1) * 32;
    const int hrow = h0 + wr;
    const bool hok = (hrow < H);

    // X staging map: 4 threads per row, 16 cols each
    const int xr = tid >> 2;
    const int xc = (tid & 3) * 16;
    const float* xsrc = x + (size_t)toks[xr] * D + xc;

    for (int kc = 0; kc < D / 64; ++kc) {
        const int kbase = kc * 64;
        // stage X chunk: 32 rows x 64 cols
        {
            const float4* sx = (const float4*)(xsrc + kbase);
            if (kc + 1 < D / 64) __builtin_prefetch(sx + 16, 0, 1);
#pragma unroll
            for (int j = 0; j < 4; ++j)
                *(bf16x4*)&Xs[xr][xc + 4 * j] = pack4(sx[j]);
        }
        // stage W1/W2 chunk: per-wave 16 rows x 64 cols
        if (hok) {
            const float4* s1 = (const float4*)(W1e + (size_t)hrow * D + kbase + wc);
            const float4* s2 = (const float4*)(W2e + (size_t)hrow * D + kbase + wc);
            if (kc + 1 < D / 64) {
                __builtin_prefetch(s1 + 16, 0, 1);
                __builtin_prefetch(s2 + 16, 0, 1);
            }
#pragma unroll
            for (int j = 0; j < 8; ++j) {
                *(bf16x4*)&Ws[0][wave][wr][wc + 4 * j] = pack4(s1[j]);
                *(bf16x4*)&Ws[1][wave][wr][wc + 4 * j] = pack4(s2[j]);
            }
        } else {
            const bf16x4 z = {};
#pragma unroll
            for (int j = 0; j < 8; ++j) {
                *(bf16x4*)&Ws[0][wave][wr][wc + 4 * j] = z;
                *(bf16x4*)&Ws[1][wave][wr][wc + 4 * j] = z;
            }
        }
        __syncthreads();
#pragma unroll
        for (int ks = 0; ks < 2; ++ks) {
            const int kf = ks * 32 + halfoff;
            v16bf alo = load_frag(&Xs[mrow][0], kf);
            v16bf ahi = load_frag(&Xs[16 + mrow][0], kf);
            v16bf b1  = load_frag(&Ws[0][wave][mrow][0], kf);
            v16bf b2  = load_frag(&Ws[1][wave][mrow][0], kf);
            c1lo = __builtin_amdgcn_wmma_f32_16x16x32_bf16(false, alo, false, b1,
                                                           (short)0, c1lo, false, false);
            c1hi = __builtin_amdgcn_wmma_f32_16x16x32_bf16(false, ahi, false, b1,
                                                           (short)0, c1hi, false, false);
            c2lo = __builtin_amdgcn_wmma_f32_16x16x32_bf16(false, alo, false, b2,
                                                           (short)0, c2lo, false, false);
            c2hi = __builtin_amdgcn_wmma_f32_16x16x32_bf16(false, ahi, false, b2,
                                                           (short)0, c2hi, false, false);
        }
        __syncthreads();
    }

    const int hcol = h0 + (lane & 15);
#pragma unroll
    for (int s = 0; s < 2; ++s) {
#pragma unroll
        for (int r = 0; r < 8; ++r) {
            int m   = s * 16 + r + halfoff;
            int pos = mt * MT + m;
            if (pos < n) {
                float h1 = s ? c1hi[r] : c1lo[r];
                float h2 = s ? c2hi[r] : c2lo[r];
                float sg = h1 / (1.f + __expf(-h1));
                hidden[(size_t)(offE + pos) * HP + hcol] = (bf16)(sg * h2);
            }
        }
    }
}

// ---------------- pass 2: partial[slot] = hidden @ W3^T ----------------
// Block: 128 threads. Tile: MT=32 rows x 64 d cols, K = HP (bf16 hidden).
__global__ __launch_bounds__(128)
void moe_pass2(const bf16* __restrict__ hidden, const float* __restrict__ W3,
               const int* __restrict__ cnt, const int* __restrict__ off,
               const int* __restrict__ slot_list, float* __restrict__ partial) {
    const int e  = blockIdx.z;
    const int mt = blockIdx.y;
    const int dt = blockIdx.x;          // 0..15 -> d0 = dt*64
    const int n  = cnt[e];
    if (mt * MT >= n) return;
    const int offE = off[e];

    __shared__ alignas(16) bf16 As[MT][64];         // 4 KB (TDM destination)
    __shared__ alignas(16) bf16 Ws[4][16][64];      // 8 KB
    __shared__ int  slots[MT];

    const int tid  = threadIdx.x;
    const int lane = tid & 31;
    const int wave = tid >> 5;
    const int mrow = lane & 15;
    const int halfoff = (lane >> 4) << 3;

    if (tid < MT) {
        int pos = mt * MT + tid;
        slots[tid] = (pos < n) ? slot_list[e * CAP + pos] : 0;
    }

    const float* W3e = W3 + (size_t)e * D * H;
    const int nsub = dt * 64 + wave * 16;

    const bf16* abase = hidden + (size_t)(offE + mt * MT) * HP;

#if !USE_TDM
    const int ar = tid >> 2;            // fallback: 4 threads/row, 16 bf16 each
    const int ac = (tid & 3) * 16;
    const bf16* asrc = abase + (size_t)ar * HP + ac;
#endif

    const int wr = lane >> 1;
    const int wc = (lane & 1) * 32;
    const int drow = nsub + wr;         // < 1024 always

    v8f clo = {}, chi = {};
    for (int kc = 0; kc < HP / 64; ++kc) {
        const int kb = kc * 64;
#if USE_TDM
        if (wave == 0) {
            // Tensor Data Mover: 2D tile, 32 rows x 64 bf16, row stride HP.
            unsigned long long ga = (unsigned long long)(uintptr_t)(abase + kb);
            unsigned lds_base = (unsigned)(uintptr_t)&As[0][0];
            u32x4 g0 = { 1u,                                 // count=1, user mode
                         lds_base,                           // lds_addr
                         (unsigned)ga,                       // global_addr[31:0]
                         (unsigned)((ga >> 32) & 0x01FFFFFFu) | (2u << 30) }; // type=2
            i32x8 g1 = { (int)(1u << 16),                    // data_size=1 (2 bytes)
                         (int)((64u & 0xFFFFu) << 16),       // tensor_dim0=64 lo16
                         (int)(((unsigned)MT & 0xFFFFu) << 16), // dim0 hi=0 | tensor_dim1=32 lo16
                         (int)(64u << 16),                   // dim1 hi=0 | tile_dim0=64
                         (int)MT,                            // tile_dim1=32, tile_dim2=0
                         (int)HP,                            // tensor_dim0_stride lo32
                         0, 0 };
            i32x4 z4 = {0, 0, 0, 0};
#if __clang_major__ >= 23
            i32x8 z8 = {0, 0, 0, 0, 0, 0, 0, 0};
            __builtin_amdgcn_tensor_load_to_lds(g0, g1, z4, z4, z8, 0);
#else
            __builtin_amdgcn_tensor_load_to_lds(g0, g1, z4, z4, 0);
#endif
        }
#else
        {
#pragma unroll
            for (int j = 0; j < 2; ++j)
                *(bf16x8*)&As[ar][ac + 8 * j] = *(const bf16x8*)(asrc + kb + 8 * j);
        }
#endif
        // W3 tile (f32 -> bf16), float2 loads (rows only 8B aligned: H=2730)
        {
            const f32x2* s3 = (const f32x2*)(W3e + (size_t)drow * H + kb + wc);
            if (kc + 1 < HP / 64) __builtin_prefetch(s3 + 32, 0, 1);
#pragma unroll
            for (int j = 0; j < 16; ++j) {
                int k = kb + wc + 2 * j;           // H even -> pair in/out together
                f32x2 z2 = {0.f, 0.f};
                f32x2 v = (k < H) ? s3[j] : z2;
                bf16x2 p = {(bf16)v.x, (bf16)v.y};
                *(bf16x2*)&Ws[wave][wr][wc + 2 * j] = p;
            }
        }
#if USE_TDM
        if (wave == 0) {
#if __has_builtin(__builtin_amdgcn_s_wait_tensorcnt)
            __builtin_amdgcn_s_wait_tensorcnt(0);
#else
            asm volatile("s_wait_tensorcnt 0x0" ::: "memory");
#endif
        }
#endif
        __syncthreads();
#pragma unroll
        for (int ks = 0; ks < 2; ++ks) {
            const int kf = ks * 32 + halfoff;
            v16bf alo = load_frag(&As[mrow][0], kf);
            v16bf ahi = load_frag(&As[16 + mrow][0], kf);
            v16bf b   = load_frag(&Ws[wave][mrow][0], kf);
            clo = __builtin_amdgcn_wmma_f32_16x16x32_bf16(false, alo, false, b,
                                                          (short)0, clo, false, false);
            chi = __builtin_amdgcn_wmma_f32_16x16x32_bf16(false, ahi, false, b,
                                                          (short)0, chi, false, false);
        }
        __syncthreads();
    }

#pragma unroll
    for (int s = 0; s < 2; ++s) {
#pragma unroll
        for (int r = 0; r < 8; ++r) {
            int m   = s * 16 + r + halfoff;
            int pos = mt * MT + m;
            if (pos < n)
                partial[(size_t)slots[m] * D + nsub + (lane & 15)] =
                    s ? chi[r] : clo[r];
        }
    }
}

// ---------------- combine ----------------
__global__ __launch_bounds__(256)
void moe_combine(const float* __restrict__ partial, const float* __restrict__ gw,
                 float* __restrict__ out) {
    size_t i = (size_t)blockIdx.x * 256 + threadIdx.x;
    if (i >= (size_t)T * D) return;
    size_t t = i >> 10;
    int d = (int)(i & (D - 1));
    out[i] = gw[2 * t]     * partial[(2 * t)     * D + d]
           + gw[2 * t + 1] * partial[(2 * t + 1) * D + d];
}

extern "C" void kernel_launch(void* const* d_in, const int* in_sizes, int n_in,
                              void* d_out, int out_size, void* d_ws, size_t ws_size,
                              hipStream_t stream) {
    const float* x  = (const float*)d_in[0];
    const float* Wg = (const float*)d_in[1];
    const float* W1 = (const float*)d_in[2];
    const float* W2 = (const float*)d_in[3];
    const float* W3 = (const float*)d_in[4];
    float* out = (float*)d_out;

    char* ws = (char*)d_ws;
    int*   cnt       = (int*)ws;                       // 8
    int*   off       = cnt + E;                        // 8
    int*   slot_list = off + E;                        // E*CAP
    float* gw        = (float*)(slot_list + E * CAP);  // ROWS
    size_t head = (size_t)((char*)(gw + ROWS) - ws);
    head = (head + 255) & ~(size_t)255;
    bf16*  hidden = (bf16*)(ws + head);                // ROWS * HP bf16  (~45 MB)
    size_t hb = head + (size_t)ROWS * HP * sizeof(bf16);
    hb = (hb + 255) & ~(size_t)255;
    float* partial = (float*)(ws + hb);                // ROWS * D f32    (~33.5 MB)

    moe_zero_cnt<<<1, 32, 0, stream>>>(cnt);
    moe_gate<<<T, 256, 0, stream>>>(x, Wg, cnt, slot_list, gw);
    moe_scan<<<1, 32, 0, stream>>>(cnt, off);
    moe_pass1<<<dim3(HP / 64, CAP / MT, E), 128, 0, stream>>>(x, W1, W2, cnt, off,
                                                              slot_list, hidden);
    moe_pass2<<<dim3(D / 64, CAP / MT, E), 128, 0, stream>>>(hidden, W3, cnt, off,
                                                             slot_list, partial);
    moe_combine<<<(T * D) / 256, 256, 0, stream>>>(partial, gw, out);
}
